// CombSubMinimumNoisedPhase_70669391889223
// MI455X (gfx1250) — matrix operand order repaired and checked by reading, not compiled, not run on hardware
//
#include <hip/hip_runtime.h>
#include <math.h>

// ---------------- constants from reference ----------------
#define SRf     44100.0f
#define BSZ     512
#define WINN    2048
#define HOP     128
#define NUNIT   256
#define BB      4
#define NFRM    512                 // NF
#define LL      262144              // NF*BS
#define PFB     1025                // WIN/2+1
#define NFR     2049                // L/HOP + 1
#define CTRLN   16528
#define KDIM    260
#define KPAD    288                 // 9 * 32
#define NTN     1033                // 16528/16
#define MT      128                 // 2048/16
#define PADL    (LL + WINN)         // 264192
#define NCH     512                 // L / BS (chunk == frame)
#define PI_F    3.14159265358979f
#define TWO_PI_F 6.28318530717959f
#define F0C     0.0083333333333333f // F0_IV/12

typedef __bf16 bf16;
typedef __attribute__((ext_vector_type(16))) __bf16 v16bf;
typedef __attribute__((ext_vector_type(2)))  __bf16 v2bf;
typedef __attribute__((ext_vector_type(8)))  float  v8f;

__device__ __forceinline__ float2 cmulf(float2 a, float2 b) {
  return make_float2(a.x*b.x - a.y*b.y, a.x*b.y + a.y*b.x);
}
__device__ __forceinline__ float hannw(int u) {
  return 0.5f - 0.5f * cosf(TWO_PI_F * (float)u / (float)WINN);
}

// ---------------- zero scratch accumulators ----------------
__global__ void k_zero(float* p, int n) {
  for (int i = blockIdx.x * blockDim.x + threadIdx.x; i < n; i += gridDim.x * blockDim.x)
    p[i] = 0.f;
}

// ---------------- pass 1: per-chunk phase-increment sums ----------------
__global__ void k_chunk_sums(const float* __restrict__ f0f, const float* __restrict__ nz,
                             float* __restrict__ csums) {
  int ch = blockIdx.x, b = blockIdx.y, t = threadIdx.x;   // 256 threads
  __shared__ float red[256];
  float f0c = f0f[b * NFRM + ch] * exp2f(-F0C);
  float acc = 0.f;
#pragma unroll
  for (int j = 0; j < 2; ++j) {
    int i = ch * BSZ + t + j * 256;
    float f0s = f0c * exp2f(nz[(size_t)b * LL + i] * F0C);
    acc += f0s / SRf;
  }
  red[t] = acc; __syncthreads();
  for (int off = 128; off > 0; off >>= 1) {
    if (t < off) red[t] += red[t + off];
    __syncthreads();
  }
  if (t == 0) csums[b * NCH + ch] = red[0];
}

// ---------------- pass 2: exclusive scan of chunk sums (wrapped mod 1) ----------------
__global__ void k_scan_chunks(const float* __restrict__ csums, float* __restrict__ cpref) {
  int b = blockIdx.x, t = threadIdx.x;                    // 512 threads
  __shared__ float s[NCH];
  float own = csums[b * NCH + t];
  s[t] = own; __syncthreads();
  for (int off = 1; off < NCH; off <<= 1) {
    float v = (t >= off) ? s[t - off] : 0.f;
    __syncthreads();
    s[t] += v;
    __syncthreads();
  }
  float ex = s[t] - own;
  ex -= roundf(ex);                                       // modular wrap
  cpref[b * NCH + t] = ex;
}

// ---------------- pass 3: per-sample phase, combtooth, frame phase sin/cos ----------------
__global__ void k_phase_comb(const float* __restrict__ f0f, const float* __restrict__ nz,
                             const float* __restrict__ cpref, float* __restrict__ comb,
                             float* __restrict__ sinp, float* __restrict__ cosp) {
  int ch = blockIdx.x, b = blockIdx.y, t = threadIdx.x;   // 512 threads
  int i = ch * BSZ + t;
  float f0c = f0f[b * NFRM + ch] * exp2f(-F0C);
  float f0s = f0c * exp2f(nz[(size_t)b * LL + i] * F0C);
  float df  = f0s / SRf;
  __shared__ float s[BSZ];
  s[t] = df; __syncthreads();
  for (int off = 1; off < BSZ; off <<= 1) {
    float v = (t >= off) ? s[t - off] : 0.f;
    __syncthreads();
    s[t] += v;
    __syncthreads();
  }
  float x = cpref[b * NCH + ch] + s[t];
  x -= roundf(x);
  float v = SRf * x / (f0s + 0.001f);
  float cb = (v == 0.f) ? 1.f : sinf(PI_F * v) / (PI_F * v);
  comb[(size_t)b * LL + i] = cb;
  if (t == 0) {
    float sn, cs;
    sincosf(TWO_PI_F * x, &sn, &cs);
    sinp[b * NFRM + ch] = sn;
    cosp[b * NFRM + ch] = cs;
  }
}

// ---------------- feats (bf16, K padded to 288) ----------------
__global__ void k_feats(const float* __restrict__ units, const float* __restrict__ f0f,
                        const float* __restrict__ vol, const float* __restrict__ sinp,
                        const float* __restrict__ cosp, bf16* __restrict__ feats) {
  int idx = blockIdx.x * blockDim.x + threadIdx.x;
  if (idx >= (BB * NFRM) * KPAD) return;
  int row = idx / KPAD, k = idx % KPAD;
  float val;
  if      (k < NUNIT)   val = units[(size_t)row * NUNIT + k];
  else if (k == NUNIT)  val = log1pf(f0f[row]);
  else if (k == NUNIT+1) val = sinp[row];
  else if (k == NUNIT+2) val = cosp[row];
  else if (k == NUNIT+3) val = vol[row];
  else                   val = 0.f;
  feats[idx] = (bf16)val;
}

// ---------------- W -> transposed bf16 (N x KPAD) ----------------
__global__ void k_wt(const float* __restrict__ W, bf16* __restrict__ Wt) {
  int idx = blockIdx.x * blockDim.x + threadIdx.x;
  if (idx >= CTRLN * KPAD) return;
  int n = idx / KPAD, k = idx % KPAD;
  Wt[idx] = (k < KDIM) ? (bf16)W[(size_t)k * CTRLN + n] : (bf16)0.f;
}

// ---------------- ctrl GEMM via v_wmma_f32_16x16x32_bf16 ----------------
__global__ void k_gemm(const bf16* __restrict__ featsb, const bf16* __restrict__ Wt,
                       const float* __restrict__ bias, float* __restrict__ ctrl) {
  __shared__ bf16 As[16 * KPAD];
  int m0 = blockIdx.x * 16;
  for (int idx = threadIdx.x; idx < 16 * KPAD; idx += 256)
    As[idx] = featsb[(size_t)(m0 + idx / KPAD) * KPAD + (idx % KPAD)];
  __syncthreads();

  int wave = threadIdx.x >> 5, lane = threadIdx.x & 31;
  int ntile = blockIdx.y * 8 + wave;
  if (ntile < NTN) {                                      // wave-uniform: EXEC stays full
    int half = lane >> 4, mn = lane & 15;
    const bf16* bp = Wt + (size_t)(ntile * 16 + mn) * KPAD;
    v8f c;
#pragma unroll
    for (int i = 0; i < 8; ++i) c[i] = 0.f;
#pragma unroll
    for (int kk = 0; kk < KPAD / 32; ++kk) {
      v16bf a;
#pragma unroll
      for (int v = 0; v < 8; ++v) {
        // A k order used here (contiguous, not ISA table): lanes 0-15 K=0..15, lanes 16-31 K=16..31 (matches B)
        int k0 = kk * 32 + (half << 4) + (v << 1);
        v2bf pr = *(const v2bf*)&As[mn * KPAD + k0];
        a[2 * v]     = pr[0];
        a[2 * v + 1] = pr[1];
      }
      // B k order used here (contiguous, same as A): lanes 0-15 K=0..15, lanes 16-31 K=16..31 (contiguous 32B run)
      v16bf bv = *(const v16bf*)(bp + kk * 32 + (half << 4));
      if (kk + 1 < KPAD / 32)
        __builtin_prefetch((const void*)(bp + (kk + 1) * 32 + (half << 4)), 0, 0);
      c = __builtin_amdgcn_wmma_f32_16x16x32_bf16(false, a, false, bv, (short)0, c,
                                                  false, false);
    }
    int col = ntile * 16 + mn;
    float bb = bias[col];
#pragma unroll
    for (int r = 0; r < 8; ++r) {
      int row = m0 + half * 8 + r;
      ctrl[(size_t)row * CTRLN + col] = c[r] + bb;
    }
  }
}

// ---------------- window-squared overlap sum ----------------
__global__ void k_wsum(float* __restrict__ wsum) {
  int pos = blockIdx.x * blockDim.x + threadIdx.x;
  if (pos >= PADL) return;
  int fmin = (pos >= WINN) ? ((pos - WINN) / HOP + 1) : 0;
  int fmax = min(NFR - 1, pos >> 7);
  float acc = 0.f;
  for (int fr = fmin; fr <= fmax; ++fr) {
    float w = hannw(pos - fr * HOP);
    acc += w * w;
  }
  wsum[pos] = acc;
}

// ---------------- in-place radix-2 DIF FFT of 1024 complex in LDS ----------------
__device__ void fft1024(float2* s, float sgn) {
  const int t = threadIdx.x;                              // 256 threads
  for (int len = 1024; len >= 2; len >>= 1) {
    int half = len >> 1;
    __syncthreads();
#pragma unroll
    for (int j = 0; j < 2; ++j) {
      int bi = t + j * 256;                               // 512 butterflies
      int blk = bi / half, pos = bi - blk * half;
      int i0 = blk * len + pos, i1 = i0 + half;
      float2 u = s[i0], v = s[i1];
      float2 d = make_float2(u.x - v.x, u.y - v.y);
      float ang = sgn * (-TWO_PI_F) * (float)pos / (float)len;
      float sn, cs; sincosf(ang, &sn, &cs);
      s[i0] = make_float2(u.x + v.x, u.y + v.y);
      s[i1] = make_float2(d.x * cs - d.y * sn, d.x * sn + d.y * cs);
    }
  }
  __syncthreads();
  float2 r[4];
#pragma unroll
  for (int j = 0; j < 4; ++j) r[j] = s[t + j * 256];
  __syncthreads();
#pragma unroll
  for (int j = 0; j < 4; ++j) {
    int i = t + j * 256;
    s[__brev((unsigned)i) >> 22] = r[j];
  }
  __syncthreads();
}

// ---------------- fused STFT * filter * minphase -> ISTFT overlap-add ----------------
__global__ void k_fft(const float* __restrict__ comb, const float* __restrict__ snoise,
                      const float* __restrict__ ctrl,
                      const float* __restrict__ mp_re,  const float* __restrict__ mp_im,
                      const float* __restrict__ mpe_re, const float* __restrict__ mpe_im,
                      float* __restrict__ hacc, float* __restrict__ nacc) {
  __shared__ float2 S[1024];
  __shared__ float2 T[1025];
  const int fr = blockIdx.x, b = blockIdx.y, which = blockIdx.z, t = threadIdx.x;
  const int base = fr * HOP - (WINN / 2);

  // windowed load, even/odd packed
  for (int n = t; n < 1024; n += 256) {
    int t0 = 2 * n, t1 = 2 * n + 1;
    int s0 = base + t0, s1 = base + t1;
    float x0 = 0.f, x1 = 0.f;
    if (which == 0) {
      if (s0 >= 0 && s0 < LL) x0 = comb[(size_t)b * LL + s0];
      if (s1 >= 0 && s1 < LL) x1 = comb[(size_t)b * LL + s1];
    } else {
      const int TLEN = WINN * 127;                        // 260096, L < 2*TLEN
      if (s0 >= 0 && s0 < LL) x0 = snoise[s0 >= TLEN ? s0 - TLEN : s0];
      if (s1 >= 0 && s1 < LL) x1 = snoise[s1 >= TLEN ? s1 - TLEN : s1];
    }
    S[n] = make_float2(x0 * hannw(t0), x1 * hannw(t1));
  }
  fft1024(S, 1.0f);

  // un-pack real spectrum, apply exp(mag)*cis(ph)*minphase
  int frc = min(fr, NFR - 2);
  int row = b * NFRM + (frc >> 2);
  int sub = frc & 3;
  size_t magBase = (size_t)row * CTRLN + (which ? 8328 : 128)  + sub * PFB;
  size_t phBase  = (size_t)row * CTRLN + (which ? 12428 : 4228) + sub * PFB;
  const float* mre = which ? mpe_re : mp_re;
  const float* mim = which ? mpe_im : mp_im;
  for (int k = t; k < 1025; k += 256) {
    float2 Zk = S[k & 1023];
    float2 Zm = S[(1024 - k) & 1023];
    float2 E = make_float2(0.5f * (Zk.x + Zm.x), 0.5f * (Zk.y - Zm.y));
    float2 O = make_float2(0.5f * (Zk.y + Zm.y), 0.5f * (Zm.x - Zk.x)); // -i/2*(Zk-conj(Zm))
    float sn, cs; sincosf(-PI_F * (float)k / 1024.f, &sn, &cs);
    float2 X = make_float2(E.x + cs * O.x - sn * O.y, E.y + cs * O.y + sn * O.x);
    float mag = expf(ctrl[magBase + k]);
    float ph  = ctrl[phBase + k];
    float ps, pc; sincosf(ph, &ps, &pc);
    float2 F = cmulf(make_float2(mag * pc, mag * ps), make_float2(mre[k], mim[k]));
    T[k] = cmulf(X, F);
  }
  __syncthreads();

  // re-pack Hermitian half-spectrum for packed inverse
  for (int k = t; k < 1024; k += 256) {
    float2 Hk = T[k];
    float2 Hm = T[1024 - k];
    float2 cHm = make_float2(Hm.x, -Hm.y);
    float2 Ze = make_float2(0.5f * (Hk.x + cHm.x), 0.5f * (Hk.y + cHm.y));
    float2 D  = make_float2(0.5f * (Hk.x - cHm.x), 0.5f * (Hk.y - cHm.y));
    float sn, cs; sincosf(PI_F * (float)k / 1024.f, &sn, &cs);
    float2 Zo = make_float2(cs * D.x - sn * D.y, cs * D.y + sn * D.x);
    S[k] = make_float2(Ze.x - Zo.y, Ze.y + Zo.x);         // Ze + i*Zo
  }
  fft1024(S, -1.0f);

  // window + overlap-add
  float* acc = (which ? nacc : hacc) + (size_t)b * PADL;
  const float inv = 1.0f / 1024.f;
  for (int n = t; n < 1024; n += 256) {
    float2 z = S[n];
    int t0 = 2 * n, t1 = 2 * n + 1;
    atomicAdd(acc + fr * HOP + t0, z.x * inv * hannw(t0));
    atomicAdd(acc + fr * HOP + t1, z.y * inv * hannw(t1));
  }
}

// ---------------- final mix: harmonic + noise*env ----------------
__global__ void k_final(const float* __restrict__ hacc, const float* __restrict__ nacc,
                        const float* __restrict__ wsum, const float* __restrict__ ctrl,
                        float* __restrict__ out) {
  int i = blockIdx.x * blockDim.x + threadIdx.x;
  if (i >= BB * LL) return;
  int b = i >> 18, ii = i & (LL - 1);
  int pos = ii + WINN / 2;
  float ws = wsum[pos] + 1e-8f;
  float h = hacc[(size_t)b * PADL + pos] / ws;
  float nz = nacc[(size_t)b * PADL + pos] / ws;
  int j = ii >> 2;                                        // NENV_DS repeat
  float env = expf(ctrl[(size_t)(b * NFRM + (j >> 7)) * CTRLN + (j & 127)]);
  out[i] = h + nz * env;
}

// ---------------- host-side orchestration ----------------
extern "C" void kernel_launch(void* const* d_in, const int* in_sizes, int n_in,
                              void* d_out, int out_size, void* d_ws, size_t ws_size,
                              hipStream_t stream) {
  const float* units  = (const float*)d_in[0];
  const float* f0f    = (const float*)d_in[1];
  const float* vol    = (const float*)d_in[2];
  const float* f0nz   = (const float*)d_in[3];
  const float* W      = (const float*)d_in[4];
  const float* bias   = (const float*)d_in[5];
  const float* snoise = (const float*)d_in[6];
  const float* mp_re  = (const float*)d_in[7];
  const float* mp_im  = (const float*)d_in[8];
  const float* mpe_re = (const float*)d_in[9];
  const float* mpe_im = (const float*)d_in[10];
  float* out = (float*)d_out;

  char* p = (char*)d_ws;
  auto take = [&](size_t bytes) { char* r = p; p += (bytes + 255) & ~(size_t)255; return r; };
  float* comb   = (float*)take((size_t)BB * LL * 4);
  float* csums  = (float*)take((size_t)BB * NCH * 4);
  float* cpref  = (float*)take((size_t)BB * NCH * 4);
  float* sinp   = (float*)take((size_t)BB * NFRM * 4);
  float* cosp   = (float*)take((size_t)BB * NFRM * 4);
  bf16*  featsb = (bf16*) take((size_t)BB * NFRM * KPAD * 2);
  bf16*  Wt     = (bf16*) take((size_t)CTRLN * KPAD * 2);
  float* ctrl   = (float*)take((size_t)BB * NFRM * CTRLN * 4);
  float* hacc   = (float*)take((size_t)BB * PADL * 4);
  float* nacc   = (float*)take((size_t)BB * PADL * 4);
  float* wsum   = (float*)take((size_t)PADL * 4);

  // zero overlap-add accumulators (hacc & nacc are contiguous)
  k_zero<<<2048, 256, 0, stream>>>(hacc, 2 * BB * PADL);

  // phase scan + combtooth
  k_chunk_sums<<<dim3(NCH, BB), 256, 0, stream>>>(f0f, f0nz, csums);
  k_scan_chunks<<<BB, NCH, 0, stream>>>(csums, cpref);
  k_phase_comb<<<dim3(NCH, BB), BSZ, 0, stream>>>(f0f, f0nz, cpref, comb, sinp, cosp);

  // control GEMM (WMMA bf16)
  k_feats<<<(BB * NFRM * KPAD + 255) / 256, 256, 0, stream>>>(units, f0f, vol, sinp, cosp, featsb);
  k_wt<<<(CTRLN * KPAD + 255) / 256, 256, 0, stream>>>(W, Wt);
  k_gemm<<<dim3(MT, (NTN + 7) / 8), 256, 0, stream>>>(featsb, Wt, bias, ctrl);

  // spectral processing
  k_wsum<<<(PADL + 255) / 256, 256, 0, stream>>>(wsum);
  k_fft<<<dim3(NFR, BB, 2), 256, 0, stream>>>(comb, snoise, ctrl, mp_re, mp_im,
                                              mpe_re, mpe_im, hacc, nacc);

  // final mix
  k_final<<<(BB * LL + 255) / 256, 256, 0, stream>>>(hacc, nacc, wsum, ctrl, out);
}
